// _ScatterMoEMLP_62388694941736
// MI455X (gfx1250) — matrix operand order repaired and checked by reading, not compiled, not run on hardware
//
#include <hip/hip_runtime.h>

// ---- problem constants (fixed by reference) ----
#define T_TOK 4096
#define HDIM  1024
#define IDIM  2048
#define NEXP  8
#define TOPK  2
#define RCAP  9216          // max padded routed rows: T*K + 8*127 rounded up to 128
#define LPITCH 40           // LDS row pitch in bf16 (32 + 8 pad) -> 80B rows, 16B aligned
#define LDSBUF (128 * LPITCH) // ushorts per buffer = 10240 B

typedef __attribute__((ext_vector_type(16))) __bf16 v16bf;
typedef __attribute__((ext_vector_type(8)))  float  v8f;

union V16U { v16bf v; uint4 q[2]; };

__device__ __forceinline__ unsigned short f2bf(float f) {
  unsigned u = __float_as_uint(f);
  u += 0x7FFFu + ((u >> 16) & 1u);   // round-to-nearest-even
  return (unsigned short)(u >> 16);
}

// LDS byte offset of a generic shared pointer (AS3 ptrtoint)
__device__ __forceinline__ unsigned lds_off(const void* p) {
  return (unsigned)(size_t)(__attribute__((address_space(3))) const void*)p;
}

__device__ __forceinline__ void async_ld_b128(unsigned lds, const void* g) {
  asm volatile("global_load_async_to_lds_b128 %0, %1, off"
               :: "v"(lds), "v"((unsigned long long)(size_t)g)
               : "memory");
}

__device__ __forceinline__ void wait_async0() {
  asm volatile("s_wait_asynccnt 0x0" ::: "memory");
}

// ---------- setup kernels ----------
__global__ void k_init(float* __restrict__ out, float* __restrict__ gates,
                       int* __restrict__ meta, int* __restrict__ ids,
                       float* __restrict__ rg) {
  int i = blockIdx.x * blockDim.x + threadIdx.x;
  if (i < T_TOK * HDIM) out[i] = 0.0f;
  if (i < T_TOK * NEXP) gates[i] = 0.0f;
  if (i < RCAP) { ids[i] = -1; rg[i] = 0.0f; }
  if (i < 32) meta[i] = 0;   // cnt[8], fill[8], base[9]
}

__global__ void k_gates(const float* __restrict__ rw, const int* __restrict__ sel,
                        float* __restrict__ gates) {
  int i = blockIdx.x * blockDim.x + threadIdx.x;
  if (i < T_TOK * TOPK) {
    int t = i / TOPK;
    int e = sel[i];
    atomicAdd(&gates[t * NEXP + e], rw[i]);
  }
}

__global__ void k_convX(const float* __restrict__ x, unsigned short* __restrict__ xb) {
  int i = blockIdx.x * blockDim.x + threadIdx.x;     // over T*H/4
  if (i < T_TOK * HDIM / 4) {
    float4 v = ((const float4*)x)[i];
    ushort4 o;
    o.x = f2bf(v.x); o.y = f2bf(v.y); o.z = f2bf(v.z); o.w = f2bf(v.w);
    ((ushort4*)xb)[i] = o;
  }
}

// in: [E][R][C] fp32  ->  out: [E][C][R] bf16
__global__ __launch_bounds__(256) void k_transpose(const float* __restrict__ in,
                                                   unsigned short* __restrict__ out,
                                                   int R, int C) {
  __shared__ float t[32][33];
  const size_t esz = (size_t)R * C;
  const float* src = in + (size_t)blockIdx.z * esz;
  unsigned short* dst = out + (size_t)blockIdx.z * esz;
  int bx = blockIdx.x * 32;                 // C index base
  int by = blockIdx.y * 32;                 // R index base
  int tx = threadIdx.x & 31, ty = threadIdx.x >> 5;
#pragma unroll
  for (int j = 0; j < 32; j += 8)
    t[ty + j][tx] = src[(size_t)(by + ty + j) * C + bx + tx];
  __syncthreads();
#pragma unroll
  for (int j = 0; j < 32; j += 8)
    dst[(size_t)(bx + ty + j) * R + by + tx] = f2bf(t[tx][ty + j]);
}

__global__ void k_count(const float* __restrict__ gates, int* __restrict__ meta) {
  int i = blockIdx.x * blockDim.x + threadIdx.x;     // over T*E
  if (i < T_TOK * NEXP && gates[i] > 0.0f) atomicAdd(&meta[i & (NEXP - 1)], 1);
}

__global__ void k_scan(int* __restrict__ meta) {
  int* cnt  = meta;
  int* base = meta + 16;
  base[0] = 0;
  for (int e = 0; e < NEXP; ++e)
    base[e + 1] = (base[e] + cnt[e] + 127) & ~127;
}

__global__ void k_fill(const float* __restrict__ gates, int* __restrict__ meta,
                       int* __restrict__ ids, float* __restrict__ rg) {
  int i = blockIdx.x * blockDim.x + threadIdx.x;     // over T*E
  if (i < T_TOK * NEXP) {
    float g = gates[i];
    if (g > 0.0f) {
      int e = i & (NEXP - 1);
      int t = i >> 3;
      int slot = atomicAdd(&meta[8 + e], 1);
      int row = meta[16 + e] + slot;
      ids[row] = t;
      rg[row]  = g;
    }
  }
}

// ---------- grouped GEMM ----------
// A:  [rows][KDIM] bf16 (FIRST: x-bf16 gathered via token ids; else h-buffer row-direct)
// BT: [E][NDIM][KDIM] bf16 (pre-transposed weights)
// FIRST: D = gelu(A*B) -> Hout bf16 [RCAP][NDIM]   (invalid rows forced to 0)
// else : atomicAdd(out[token], gate * (A*B))
template <int KDIM, int NDIM, bool FIRST>
__global__ __launch_bounds__(256) void k_gemm(const unsigned short* __restrict__ A,
                                              const unsigned short* __restrict__ BT,
                                              const int* __restrict__ ids,
                                              const float* __restrict__ rg,
                                              const int* __restrict__ base,
                                              unsigned short* __restrict__ Hout,
                                              float* __restrict__ Out) {
  const int m0 = blockIdx.y * 128;
  const int n0 = blockIdx.x * 128;
  const int ptot = base[NEXP];
  if (m0 >= ptot) return;

  int e = 0;
#pragma unroll
  for (int i = 1; i <= NEXP; ++i) if (m0 >= base[i]) e = i;   // m0 < base[8] => e <= 7
  const unsigned short* Bexp = BT + (size_t)e * NDIM * KDIM;

  __shared__ unsigned short Al[2 * LDSBUF];
  __shared__ unsigned short Bl[2 * LDSBUF];

  const int tid  = threadIdx.x;
  const int lane = tid & 31;
  const int wv   = tid >> 5;
  const int wm   = wv & 3;        // 4 waves along M (32 rows each)
  const int wn   = wv >> 2;       // 2 waves along N (64 cols each)
  const int hh   = lane >> 4;     // lane half
  const int ln   = lane & 15;

  // ---- per-thread tile-load geometry (chunk c = tid + j*256; row=c>>2, q=c&3) ----
  const int row0 = tid >> 2,          q0 = (tid & 3) * 8;       // rows 0..63
  const int row1 = (tid + 256) >> 2,  q1 = q0;                  // rows 64..127
  // A source rows (token-gathered for pass 1; clamp invalid to 0 - epilogue masks)
  int t0, t1;
  if (FIRST) {
    int i0 = ids[m0 + row0]; t0 = i0 < 0 ? 0 : i0;
    int i1 = ids[m0 + row1]; t1 = i1 < 0 ? 0 : i1;
  } else {
    t0 = m0 + row0; t1 = m0 + row1;
  }
  const unsigned short* pA0 = A + (size_t)t0 * KDIM + q0;
  const unsigned short* pA1 = A + (size_t)t1 * KDIM + q1;
  const unsigned short* pB0 = Bexp + (size_t)(n0 + row0) * KDIM + q0;
  const unsigned short* pB1 = Bexp + (size_t)(n0 + row1) * KDIM + q1;
  const unsigned lA0 = lds_off(&Al[row0 * LPITCH + q0]);
  const unsigned lA1 = lds_off(&Al[row1 * LPITCH + q1]);
  const unsigned lB0 = lds_off(&Bl[row0 * LPITCH + q0]);
  const unsigned lB1 = lds_off(&Bl[row1 * LPITCH + q1]);

  v8f acc[2][4];
#pragma unroll
  for (int a = 0; a < 2; ++a)
#pragma unroll
    for (int b = 0; b < 4; ++b)
#pragma unroll
      for (int r = 0; r < 8; ++r) acc[a][b][r] = 0.0f;

  // issue 4 async global->LDS b128 copies into buffer BUF, advance K by 32
  auto stage = [&](unsigned bo) {
    async_ld_b128(lA0 + bo, pA0);
    async_ld_b128(lA1 + bo, pA1);
    async_ld_b128(lB0 + bo, pB0);
    async_ld_b128(lB1 + bo, pB1);
    pA0 += 32; pA1 += 32; pB0 += 32; pB1 += 32;   // next 32-K slab (64 B)
  };

  const int ra0 = (wm * 32 + ln) * LPITCH;          // A frag rows (ms stride 16*LPITCH)
  const int rb0 = (wn * 64 + ln) * LPITCH;          // B frag rows (ns stride 16*LPITCH)
  auto compute = [&](const unsigned short* Ab, const unsigned short* Bb) {
    V16U a[2], b[4];
#pragma unroll
    for (int ms = 0; ms < 2; ++ms) {
      int r = ra0 + ms * 16 * LPITCH;
      a[ms].q[0] = *(const uint4*)&Ab[r + hh * 8];        // K = h*8 .. +7
      a[ms].q[1] = *(const uint4*)&Ab[r + 16 + hh * 8];   // K = 16+h*8 .. +7
    }
#pragma unroll
    for (int ns = 0; ns < 4; ++ns) {
      int r = rb0 + ns * 16 * LPITCH;
      b[ns].q[0] = *(const uint4*)&Bb[r + hh * 8];        // K = h*8 .. +7
      b[ns].q[1] = *(const uint4*)&Bb[r + 16 + hh * 8];   // K = 16+h*8 .. +7
    }
#pragma unroll
    for (int ms = 0; ms < 2; ++ms)
#pragma unroll
      for (int ns = 0; ns < 4; ++ns)
        acc[ms][ns] = __builtin_amdgcn_wmma_f32_16x16x32_bf16(
            false, a[ms].v, false, b[ns].v, (short)0, acc[ms][ns], false, false);
  };

  constexpr int NK = KDIM / 32;                 // 32 or 64: even
  constexpr unsigned B1 = (unsigned)(LDSBUF * 2);  // byte offset of buffer 1
  stage(0u);
  wait_async0();
  __syncthreads();
#pragma unroll 1
  for (int kt = 0; kt < NK - 2; kt += 2) {
    stage(B1);
    compute(&Al[0], &Bl[0]);
    wait_async0();
    __syncthreads();
    stage(0u);
    compute(&Al[LDSBUF], &Bl[LDSBUF]);
    wait_async0();
    __syncthreads();
  }
  stage(B1);
  compute(&Al[0], &Bl[0]);
  wait_async0();
  __syncthreads();
  compute(&Al[LDSBUF], &Bl[LDSBUF]);

  if (FIRST) {
#pragma unroll
    for (int ms = 0; ms < 2; ++ms) {
#pragma unroll
      for (int r = 0; r < 8; ++r) {
        int gm = m0 + wm * 32 + ms * 16 + r + 8 * hh;
        bool valid = ids[gm] >= 0;
#pragma unroll
        for (int ns = 0; ns < 4; ++ns) {
          int gn = n0 + wn * 64 + ns * 16 + ln;
          float x = acc[ms][ns][r];
          float g = valid ? 0.5f * x * (1.0f + erff(x * 0.70710678118654752f)) : 0.0f;
          Hout[(size_t)gm * NDIM + gn] = f2bf(g);
        }
      }
    }
  } else {
#pragma unroll
    for (int ms = 0; ms < 2; ++ms) {
#pragma unroll
      for (int r = 0; r < 8; ++r) {
        int gm = m0 + wm * 32 + ms * 16 + r + 8 * hh;
        int tok = ids[gm];
        if (tok >= 0) {
          float gate = rg[gm];
#pragma unroll
          for (int ns = 0; ns < 4; ++ns) {
            int gn = n0 + wn * 64 + ns * 16 + ln;
            atomicAdd(&Out[(size_t)tok * NDIM + gn], gate * acc[ms][ns][r]);
          }
        }
      }
    }
  }
}

extern "C" void kernel_launch(void* const* d_in, const int* in_sizes, int n_in,
                              void* d_out, int out_size, void* d_ws, size_t ws_size,
                              hipStream_t stream) {
  const float* x   = (const float*)d_in[0];
  const float* rw  = (const float*)d_in[1];
  const int*   sel = (const int*)d_in[2];
  const float* wfc = (const float*)d_in[3];
  const float* wpj = (const float*)d_in[4];
  float* out = (float*)d_out;

  char* ws = (char*)d_ws;
  size_t off = 0;
  auto alloc = [&](size_t bytes) -> void* {
    void* p = ws + off;
    off = (off + bytes + 255) & ~(size_t)255;
    return p;
  };
  float* gates = (float*)alloc((size_t)T_TOK * NEXP * 4);
  int*   meta  = (int*)alloc(32 * 4);                       // cnt[8] fill[8] base[9]
  int*   ids   = (int*)alloc((size_t)RCAP * 4);
  float* rgv   = (float*)alloc((size_t)RCAP * 4);
  unsigned short* Xbf  = (unsigned short*)alloc((size_t)T_TOK * HDIM * 2);
  unsigned short* WfcT = (unsigned short*)alloc((size_t)NEXP * HDIM * IDIM * 2);
  unsigned short* WpjT = (unsigned short*)alloc((size_t)NEXP * HDIM * IDIM * 2);
  unsigned short* Hbuf = (unsigned short*)alloc((size_t)RCAP * IDIM * 2);

  k_init <<<(T_TOK * HDIM + 255) / 256, 256, 0, stream>>>(out, gates, meta, ids, rgv);
  k_gates<<<(T_TOK * TOPK + 255) / 256, 256, 0, stream>>>(rw, sel, gates);
  k_convX<<<(T_TOK * HDIM / 4 + 255) / 256, 256, 0, stream>>>(x, Xbf);
  k_transpose<<<dim3(IDIM / 32, HDIM / 32, NEXP), 256, 0, stream>>>(wfc, WfcT, HDIM, IDIM);
  k_transpose<<<dim3(HDIM / 32, IDIM / 32, NEXP), 256, 0, stream>>>(wpj, WpjT, IDIM, HDIM);
  k_count<<<(T_TOK * NEXP + 255) / 256, 256, 0, stream>>>(gates, meta);
  k_scan <<<1, 1, 0, stream>>>(meta);
  k_fill <<<(T_TOK * NEXP + 255) / 256, 256, 0, stream>>>(gates, meta, ids, rgv);

  k_gemm<HDIM, IDIM, true ><<<dim3(IDIM / 128, RCAP / 128), 256, 0, stream>>>(
      Xbf, WfcT, ids, rgv, meta + 16, Hbuf, nullptr);
  k_gemm<IDIM, HDIM, false><<<dim3(HDIM / 128, RCAP / 128), 256, 0, stream>>>(
      Hbuf, WpjT, ids, rgv, meta + 16, nullptr, out);
}
